// DiTAttention_27685359190164
// MI455X (gfx1250) — matrix operand order repaired and checked by reading, not compiled, not run on hardware
//
#include <hip/hip_runtime.h>
#include <cstdint>
#include <cstddef>

// ---------------------------------------------------------------------------
// DiT block stack for MI455X (gfx1250, wave32, WMMA).
// ~283 GFLOP total, ~100MB/layer streamed -> matrix-pipe bound.
// All GEMMs + attention run through v_wmma_f32_16x16x32_f16 (f32 accum).
// GEMM: 128x128x32 tiles, ping-pong LDS double buffering, packed-b32 B staging.
// ---------------------------------------------------------------------------

constexpr int CB = 8;          // batch
constexpr int CN = 1024;       // seq len
constexpr int CD = 512;        // model dim
constexpr int CH = 8;          // heads
constexpr int CHD = 64;        // head dim
constexpr int CL = 6;          // layers
constexpr int CM = CB * CN;    // 8192 rows

typedef __attribute__((ext_vector_type(16))) _Float16 v16h;
typedef __attribute__((ext_vector_type(8)))  float    v8f;

struct __align__(16) U4 { unsigned int x, y, z, w; };
union AF { U4 u[2]; v16h v; };   // one WMMA 16x32 (A) or 32x16 (B) f16 fragment
union PK { _Float16 h[2]; unsigned int w; };

__device__ __forceinline__ v8f wmma_f16(const AF& a, const AF& b, v8f c) {
  return __builtin_amdgcn_wmma_f32_16x16x32_f16(false, a.v, false, b.v,
                                                (short)0, c, false, false);
}

__device__ __forceinline__ float redmax16(float v) {
#pragma unroll
  for (int o = 1; o < 16; o <<= 1) v = fmaxf(v, __shfl_xor(v, o, 32));
  return v;
}
__device__ __forceinline__ float redsum16(float v) {
#pragma unroll
  for (int o = 1; o < 16; o <<= 1) v += __shfl_xor(v, o, 32);
  return v;
}
__device__ __forceinline__ float fast_tanh(float x) {
  float e = __expf(2.f * x);
  return (e - 1.f) / (e + 1.f);
}

__device__ __forceinline__ float block_reduce_sum(float v, float* sbuf) {
#pragma unroll
  for (int o = 16; o > 0; o >>= 1) v += __shfl_xor(v, o, 32);
  int wid = threadIdx.x >> 5, lane = threadIdx.x & 31;
  if (lane == 0) sbuf[wid] = v;
  __syncthreads();
  if (wid == 0) {
    float t = (lane < 8) ? sbuf[lane] : 0.f;
#pragma unroll
    for (int o = 4; o > 0; o >>= 1) t += __shfl_xor(t, o, 32);
    if (lane == 0) sbuf[0] = t;
  }
  __syncthreads();
  v = sbuf[0];
  __syncthreads();
  return v;
}

// ---------------------------------------------------------------------------
// timestep MLP: c = silu(temb(t) @ tw1 + tb1) @ tw2 + tb2
// ---------------------------------------------------------------------------
__global__ __launch_bounds__(256) void tmlp_kernel(
    const float* __restrict__ t, const float* __restrict__ tw1,
    const float* __restrict__ tb1, const float* __restrict__ tw2,
    const float* __restrict__ tb2, float* __restrict__ c) {
  int b = blockIdx.x;
  float tv = t[b];
  __shared__ float emb[256];
  __shared__ float c1[512];
  int tid = threadIdx.x;
  {
    int jj = tid & 127;
    float fr = __expf(-9.210340371976184f * (float)jj / 128.f);  // ln(1e4)
    float arg = tv * fr;
    emb[tid] = (tid < 128) ? __cosf(arg) : __sinf(arg);
  }
  __syncthreads();
  for (int d = tid; d < 512; d += 256) {
    float s = tb1[d];
#pragma unroll 8
    for (int k = 0; k < 256; ++k) s += emb[k] * tw1[k * 512 + d];
    c1[d] = s / (1.f + __expf(-s));           // silu
  }
  __syncthreads();
  for (int d = tid; d < 512; d += 256) {
    float s = tb2[d];
#pragma unroll 8
    for (int k = 0; k < 512; ++k) s += c1[k] * tw2[k * 512 + d];
    c[b * 512 + d] = s;
  }
}

// z = x + (x + pe) = 2x + pe
__global__ __launch_bounds__(256) void initz_kernel(
    const float* __restrict__ x, const float* __restrict__ pe,
    float* __restrict__ z) {
  int idx = blockIdx.x * 256 + threadIdx.x;
  z[idx] = 2.f * x[idx] + pe[idx & ((CN * CD) - 1)];
}

// ada[b, j] = silu(c[b]) @ Wada[:, j] + bada[j]
__global__ __launch_bounds__(256) void ada_kernel(
    const float* __restrict__ c, const float* __restrict__ Wada,
    const float* __restrict__ bada, float* __restrict__ ada) {
  int b = blockIdx.y;
  int j = blockIdx.x * 256 + threadIdx.x;          // 0..3071
  __shared__ float sc[512];
  for (int k = threadIdx.x; k < 512; k += 256) {
    float v = c[b * 512 + k];
    sc[k] = v / (1.f + __expf(-v));
  }
  __syncthreads();
  float s = bada[j];
#pragma unroll 8
  for (int k = 0; k < 512; ++k) s += sc[k] * Wada[(size_t)k * 3072 + j];
  ada[b * 3072 + j] = s;
}

// h = layernorm(z) * (1 + sc) + sh  -> f16 (GEMM A operand)
__global__ __launch_bounds__(256) void ln_mod_kernel(
    const float* __restrict__ z, const float* __restrict__ ada,
    int sh_off, int sc_off, _Float16* __restrict__ hb) {
  __shared__ float sbuf[8];
  int row = blockIdx.x, b = row >> 10;
  const float* zr = z + (size_t)row * 512;
  int tid = threadIdx.x;
  float x0 = zr[tid], x1 = zr[tid + 256];
  float mean = block_reduce_sum(x0 + x1, sbuf) * (1.f / 512.f);
  float d0 = x0 - mean, d1 = x1 - mean;
  float var = block_reduce_sum(d0 * d0 + d1 * d1, sbuf) * (1.f / 512.f);
  float rinv = rsqrtf(var + 1e-6f);
  const float* ab = ada + b * 3072;
  hb[(size_t)row * 512 + tid] =
      (_Float16)(d0 * rinv * (1.f + ab[sc_off + tid]) + ab[sh_off + tid]);
  hb[(size_t)row * 512 + tid + 256] =
      (_Float16)(d1 * rinv * (1.f + ab[sc_off + tid + 256]) + ab[sh_off + tid + 256]);
}

// ---------------------------------------------------------------------------
// WMMA GEMM: C[M,Nn] = act(A_f16[M,K] @ W_f32[K,Nn] + bias)
// 128x128x32 block tile, 8 waves of 32x64, ping-pong LDS double buffering.
// B staged transposed ([n][k], k-pairs packed to dwords) so B-fragment loads
// are contiguous ds_load_b128 per the CDNA5 32x16 B layout.
// bmode: 0 = row-major f16 store, 1 = [B,H,N,HD] head layout, 2 = [B,H,HD,N].
// ---------------------------------------------------------------------------
#define TLD 40   // LDS row stride (f16 elems): keeps 16B alignment (80B rows)
__global__ __launch_bounds__(256) void gemm_kernel(
    const _Float16* __restrict__ A, const float* __restrict__ W,
    const float* __restrict__ bias, float* __restrict__ Cf,
    _Float16* __restrict__ Cb, int M, int Nn, int K, int act, int bmode) {
  __shared__ __align__(16) _Float16 As[2][128 * TLD];
  __shared__ __align__(16) _Float16 Bs[2][128 * TLD];
  int tid = threadIdx.x;
  int lane = tid & 31, wid = tid >> 5;
  int hf = lane >> 4, ln = lane & 15;
  int wm = wid & 3, wn = wid >> 2;
  int rowBase = blockIdx.y * 128;
  int colBase = blockIdx.x * 128;

  v8f acc[2][4];
  v8f zz = {0.f, 0.f, 0.f, 0.f, 0.f, 0.f, 0.f, 0.f};
#pragma unroll
  for (int i = 0; i < 2; ++i)
#pragma unroll
    for (int j = 0; j < 4; ++j) acc[i][j] = zz;

  int arow = tid >> 1, acol = (tid & 1) * 16;  // A: 16 f16 per thread
  int kp = tid >> 4, bn0 = (tid & 15) * 8;     // B: k-pair 2kp..2kp+1, 8 n

  // staging registers (global -> regs -> LDS pipeline)
  U4 ra0, ra1;
  float4 rb0, rb1, rb2, rb3;

  auto gload = [&](int kk) {
    const _Float16* ag = A + (size_t)(rowBase + arow) * K + kk + acol;
    ra0 = *(const U4*)ag;
    ra1 = *(const U4*)(ag + 8);
    const float* wg = W + (size_t)(kk + 2 * kp) * Nn + colBase + bn0;
    rb0 = *(const float4*)wg;
    rb1 = *(const float4*)(wg + 4);
    rb2 = *(const float4*)(wg + Nn);
    rb3 = *(const float4*)(wg + Nn + 4);
  };
  auto sstore = [&](int buf) {
    *(U4*)&As[buf][arow * TLD + acol] = ra0;
    *(U4*)&As[buf][arow * TLD + acol + 8] = ra1;
    float lo[8] = {rb0.x, rb0.y, rb0.z, rb0.w, rb1.x, rb1.y, rb1.z, rb1.w};
    float hi[8] = {rb2.x, rb2.y, rb2.z, rb2.w, rb3.x, rb3.y, rb3.z, rb3.w};
#pragma unroll
    for (int j = 0; j < 8; ++j) {
      PK p;
      p.h[0] = (_Float16)lo[j];   // k = 2kp
      p.h[1] = (_Float16)hi[j];   // k = 2kp+1
      *(unsigned int*)&Bs[buf][(bn0 + j) * TLD + 2 * kp] = p.w;
    }
  };

  gload(0);
  sstore(0);
  __syncthreads();

  int buf = 0;
  for (int kk = 0; kk < K; kk += 32) {
    bool more = (kk + 32) < K;
    if (more) gload(kk + 32);                          // overlap with compute
    if (kk + 64 < K)
      __builtin_prefetch(A + (size_t)(rowBase + arow) * K + kk + 64 + acol, 0, 1);

    // load all fragments up front so the 8 WMMAs can issue back-to-back
    AF a[2];
#pragma unroll
    for (int i = 0; i < 2; ++i) {
      // A 16x32 layout: lane<16 holds K {0..7,16..23}, lane>=16 {8..15,24..31}
      const _Float16* ap = &As[buf][(wm * 32 + i * 16 + ln) * TLD + hf * 8];
      a[i].u[0] = *(const U4*)ap;
      a[i].u[1] = *(const U4*)(ap + 16);
    }
    AF bb[4];
#pragma unroll
    for (int j = 0; j < 4; ++j) {
      // B 32x16 layout: lane holds column n=ln, lane<16 K {0..7,16..23}, lane>=16 {8..15,24..31}
      const _Float16* bp = &Bs[buf][(wn * 64 + j * 16 + ln) * TLD + hf * 8];
      bb[j].u[0] = *(const U4*)bp;
      bb[j].u[1] = *(const U4*)(bp + 16);
    }
#pragma unroll
    for (int j = 0; j < 4; ++j)
#pragma unroll
      for (int i = 0; i < 2; ++i) acc[i][j] = wmma_f16(a[i], bb[j], acc[i][j]);

    if (more) sstore(buf ^ 1);   // other waves still read buf; disjoint buffer
    __syncthreads();
    buf ^= 1;
  }

  // epilogue (C layout: VGPR r, lane l -> M = r + 8*(l>>4), N = l&15)
#pragma unroll
  for (int i = 0; i < 2; ++i)
#pragma unroll
    for (int j = 0; j < 4; ++j) {
      int gcol = colBase + wn * 64 + j * 16 + ln;
      float bv = bias[gcol];
#pragma unroll
      for (int r = 0; r < 8; ++r) {
        int grow = rowBase + wm * 32 + i * 16 + r + 8 * hf;
        float val = acc[i][j][r] + bv;
        if (act == 1) {  // tanh-approx gelu
          float tg = 0.7978845608028654f * (val + 0.044715f * val * val * val);
          val = 0.5f * val * (1.f + fast_tanh(tg));
        }
        if (Cf) Cf[(size_t)grow * Nn + gcol] = val;
        if (Cb) {
          size_t idx;
          if (bmode == 0) {
            idx = (size_t)grow * Nn + gcol;
          } else {
            int b = grow >> 10, n = grow & 1023;
            int h = gcol >> 6, e = gcol & 63;
            if (bmode == 1) idx = (((size_t)(b * CH + h)) * CN + n) * CHD + e;
            else            idx = (((size_t)(b * CH + h)) * CHD + e) * CN + n;
          }
          Cb[idx] = (_Float16)val;
        }
      }
    }
}

// ---------------------------------------------------------------------------
// Flash attention: grid (B*H, N/128); 8 waves x 16 query rows.
// Keys streamed in chunks of 32 (= WMMA K).  Online softmax, P bounced
// through wave-private LDS (C-layout -> A-layout), ctx += P @ V(transposed).
// ---------------------------------------------------------------------------
__global__ __launch_bounds__(256) void attn_kernel(
    const _Float16* __restrict__ qh, const _Float16* __restrict__ kh,
    const _Float16* __restrict__ vt, const float* __restrict__ mask,
    float* __restrict__ out) {
  int bh = blockIdx.x;
  int b = bh >> 3, h = bh & 7;
  int wid = threadIdx.x >> 5, lane = threadIdx.x & 31;
  int hf = lane >> 4, ln = lane & 15;
  int qrow0 = blockIdx.y * 128 + wid * 16;

  const _Float16* qp = qh + ((size_t)bh * CN + qrow0) * CHD;
  const _Float16* kp = kh + (size_t)bh * CN * CHD;
  const _Float16* vp = vt + (size_t)bh * CHD * CN;

  AF a0, a1;  // Q fragments: kd 0..31 and 32..63
  {
    const _Float16* qr = qp + ln * CHD + hf * 8;
    a0.u[0] = *(const U4*)qr;        a0.u[1] = *(const U4*)(qr + 16);
    a1.u[0] = *(const U4*)(qr + 32); a1.u[1] = *(const U4*)(qr + 48);
  }
  float mq[8], mrun[8], lrun[8];
#pragma unroll
  for (int r = 0; r < 8; ++r) {
    mq[r] = mask[b * CN + qrow0 + r + 8 * hf];
    mrun[r] = -1e30f;
    lrun[r] = 0.f;
  }
  v8f zz = {0.f, 0.f, 0.f, 0.f, 0.f, 0.f, 0.f, 0.f};
  v8f acc[4] = {zz, zz, zz, zz};

  __shared__ __align__(16) _Float16 P[8][16 * 32];  // wave-private P buffers
  _Float16* pl = P[wid];

  for (int c0 = 0; c0 < CN; c0 += 32) {
    AF bk0[2], bk1[2];  // [kdim-half] for key tiles 0 and 1
#pragma unroll
    for (int t = 0; t < 2; ++t) {
      const _Float16* kr = kp + (size_t)(c0 + t * 16 + ln) * CHD + hf * 8;
      AF p0, p1;
      p0.u[0] = *(const U4*)kr;        p0.u[1] = *(const U4*)(kr + 16);
      p1.u[0] = *(const U4*)(kr + 32); p1.u[1] = *(const U4*)(kr + 48);
      if (t == 0) { bk0[0] = p0; bk0[1] = p1; }
      else        { bk1[0] = p0; bk1[1] = p1; }
    }
    v8f s0 = zz, s1 = zz;
    s0 = wmma_f16(a0, bk0[0], s0); s0 = wmma_f16(a1, bk0[1], s0);
    s1 = wmma_f16(a0, bk1[0], s1); s1 = wmma_f16(a1, bk1[1], s1);

    float mk0 = mask[b * CN + c0 + ln];
    float mk1 = mask[b * CN + c0 + 16 + ln];
#pragma unroll
    for (int r = 0; r < 8; ++r) {
      float mm0 = mq[r] * mk0; mm0 = (mm0 == 0.f) ? -10000.f : mm0;
      float mm1 = mq[r] * mk1; mm1 = (mm1 == 0.f) ? -10000.f : mm1;
      float v0 = s0[r] * 0.125f + mm0;     // 1/sqrt(64)
      float v1 = s1[r] * 0.125f + mm1;
      float cm = redmax16(fmaxf(v0, v1));
      float nm = fmaxf(mrun[r], cm);
      float corr = __expf(mrun[r] - nm);
      mrun[r] = nm;
      float p0 = __expf(v0 - nm), p1 = __expf(v1 - nm);
      lrun[r] = lrun[r] * corr + redsum16(p0 + p1);
      acc[0][r] *= corr; acc[1][r] *= corr;
      acc[2][r] *= corr; acc[3][r] *= corr;
      pl[(r + 8 * hf) * 32 + ln] = (_Float16)p0;
      pl[(r + 8 * hf) * 32 + 16 + ln] = (_Float16)p1;
    }
    // reload P as A-fragment (rows 0..15, K = 32 chunk keys)
    AF pf;
    const _Float16* pr = pl + ln * 32 + hf * 8;
    pf.u[0] = *(const U4*)pr;
    pf.u[1] = *(const U4*)(pr + 16);
#pragma unroll
    for (int j = 0; j < 4; ++j) {
      AF bv;
      const _Float16* vr = vp + (size_t)(j * 16 + ln) * CN + c0 + hf * 8;
      bv.u[0] = *(const U4*)vr;
      bv.u[1] = *(const U4*)(vr + 16);
      acc[j] = wmma_f16(pf, bv, acc[j]);
    }
  }
#pragma unroll
  for (int r = 0; r < 8; ++r) {
    float inv = 1.f / lrun[r];
    size_t rowo = ((size_t)b * CN + qrow0 + r + 8 * hf) * CD + h * CHD;
#pragma unroll
    for (int j = 0; j < 4; ++j) out[rowo + j * 16 + ln] = acc[j][r] * inv;
  }
}

// z += ada[b, off + d] * val
__global__ __launch_bounds__(256) void gate_add_kernel(
    float* __restrict__ z, const float* __restrict__ ada, int off,
    const float* __restrict__ val) {
  int idx = blockIdx.x * 256 + threadIdx.x;
  int d = idx & (CD - 1);
  int b = idx >> 19;                     // N*D = 2^19
  z[idx] += ada[b * 3072 + off + d] * val[idx];
}

// lorentz projx in-place: z[...,0] = sqrt(1 + sum(space^2))
__global__ __launch_bounds__(256) void projx_kernel(float* __restrict__ z) {
  __shared__ float sbuf[8];
  float* zr = z + (size_t)blockIdx.x * CD;
  int tid = threadIdx.x;
  float a = (tid == 0) ? 0.f : zr[tid];
  float c = zr[tid + 256];
  float s = block_reduce_sum(a * a + c * c, sbuf);
  if (tid == 0) zr[0] = sqrtf(1.0f + s);
}

// final projx + mask-select into d_out
__global__ __launch_bounds__(256) void finalize_kernel(
    const float* __restrict__ z, const float* __restrict__ mask,
    float* __restrict__ out) {
  __shared__ float sbuf[8];
  int row = blockIdx.x;
  const float* zr = z + (size_t)row * CD;
  int tid = threadIdx.x;
  float a = (tid == 0) ? 0.f : zr[tid];
  float c = zr[tid + 256];
  float s = block_reduce_sum(a * a + c * c, sbuf);
  float mk = mask[row];
  float x0 = sqrtf(1.0f + s);
  float* orow = out + (size_t)row * CD;
  orow[tid] = (mk != 0.f) ? ((tid == 0) ? x0 : zr[tid]) : 0.f;
  orow[tid + 256] = (mk != 0.f) ? zr[tid + 256] : 0.f;
}

// ---------------------------------------------------------------------------
extern "C" void kernel_launch(void* const* d_in, const int* in_sizes, int n_in,
                              void* d_out, int out_size, void* d_ws,
                              size_t ws_size, hipStream_t stream) {
  (void)in_sizes; (void)n_in; (void)out_size; (void)ws_size;
  const float* t    = (const float*)d_in[0];
  const float* x    = (const float*)d_in[1];
  const float* mask = (const float*)d_in[2];
  const float* pe   = (const float*)d_in[3];
  const float* tw1  = (const float*)d_in[4];
  const float* tb1  = (const float*)d_in[5];
  const float* tw2  = (const float*)d_in[6];
  const float* tb2  = (const float*)d_in[7];
  const float* Wq   = (const float*)d_in[8];
  const float* bq   = (const float*)d_in[9];
  const float* Wk   = (const float*)d_in[10];
  const float* bk   = (const float*)d_in[11];
  const float* Wv   = (const float*)d_in[12];
  const float* bv   = (const float*)d_in[13];
  const float* W1   = (const float*)d_in[14];
  const float* b1   = (const float*)d_in[15];
  const float* W2   = (const float*)d_in[16];
  const float* b2   = (const float*)d_in[17];
  const float* Wada = (const float*)d_in[18];
  const float* bada = (const float*)d_in[19];
  float* out = (float*)d_out;

  char* ws = (char*)d_ws;
  size_t off = 0;
  auto carve = [&](size_t bytes) -> char* {
    char* p = ws + off;
    off += (bytes + 255) & ~(size_t)255;
    return p;
  };
  float* c_      = (float*)carve((size_t)CB * CD * 4);          // 16 KB
  float* ada_    = (float*)carve((size_t)CB * 6 * CD * 4);      // 96 KB
  float* z_      = (float*)carve((size_t)CM * CD * 4);          // 16 MB
  _Float16* hh   = (_Float16*)carve((size_t)CM * CD * 2);       // 8 MB
  char* head     = carve((size_t)CM * 4 * CD * 2);              // 32 MB
  _Float16* qh   = (_Float16*)head;                             // [B,H,N,HD]
  _Float16* kh   = (_Float16*)(head + (size_t)CM * CD * 2);     // [B,H,N,HD]
  _Float16* vt   = (_Float16*)(head + 2 * (size_t)CM * CD * 2); // [B,H,HD,N]
  _Float16* mb   = (_Float16*)head;                             // MLP hidden
  float* tmp     = (float*)carve((size_t)CM * CD * 4);          // 16 MB

  tmlp_kernel<<<dim3(CB), dim3(256), 0, stream>>>(t, tw1, tb1, tw2, tb2, c_);
  initz_kernel<<<dim3((CM * CD) / 256), dim3(256), 0, stream>>>(x, pe, z_);

  for (int i = 0; i < CL; ++i) {
    ada_kernel<<<dim3(12, CB), dim3(256), 0, stream>>>(
        c_, Wada + (size_t)i * CD * 6 * CD, bada + (size_t)i * 6 * CD, ada_);

    // --- attention branch ---
    ln_mod_kernel<<<dim3(CM), dim3(256), 0, stream>>>(z_, ada_, 0, CD, hh);
    size_t wofs = (size_t)i * CD * CD;
    gemm_kernel<<<dim3(CD / 128, CM / 128), dim3(256), 0, stream>>>(
        hh, Wq + wofs, bq + (size_t)i * CD, nullptr, qh, CM, CD, CD, 0, 1);
    gemm_kernel<<<dim3(CD / 128, CM / 128), dim3(256), 0, stream>>>(
        hh, Wk + wofs, bk + (size_t)i * CD, nullptr, kh, CM, CD, CD, 0, 1);
    gemm_kernel<<<dim3(CD / 128, CM / 128), dim3(256), 0, stream>>>(
        hh, Wv + wofs, bv + (size_t)i * CD, nullptr, vt, CM, CD, CD, 0, 2);
    attn_kernel<<<dim3(CB * CH, CN / 128), dim3(256), 0, stream>>>(
        qh, kh, vt, mask, tmp);
    gate_add_kernel<<<dim3((CM * CD) / 256), dim3(256), 0, stream>>>(
        z_, ada_, 2 * CD, tmp);
    projx_kernel<<<dim3(CM), dim3(256), 0, stream>>>(z_);

    // --- MLP branch ---
    ln_mod_kernel<<<dim3(CM), dim3(256), 0, stream>>>(z_, ada_, 3 * CD, 4 * CD, hh);
    gemm_kernel<<<dim3((4 * CD) / 128, CM / 128), dim3(256), 0, stream>>>(
        hh, W1 + (size_t)i * CD * 4 * CD, b1 + (size_t)i * 4 * CD,
        nullptr, mb, CM, 4 * CD, CD, 1, 0);
    gemm_kernel<<<dim3(CD / 128, CM / 128), dim3(256), 0, stream>>>(
        mb, W2 + (size_t)i * 4 * CD * CD, b2 + (size_t)i * CD,
        tmp, nullptr, CM, CD, 4 * CD, 0, 0);
    gate_add_kernel<<<dim3((CM * CD) / 256), dim3(256), 0, stream>>>(
        z_, ada_, 5 * CD, tmp);
    projx_kernel<<<dim3(CM), dim3(256), 0, stream>>>(z_);
  }
  finalize_kernel<<<dim3(CM), dim3(256), 0, stream>>>(z_, mask, out);
}
